// SupervisedGraphsage_26963804685008
// MI455X (gfx1250) — matrix operand order repaired and checked by reading, not compiled, not run on hardware
//
#include <hip/hip_runtime.h>
#include <stdint.h>

// ---------------------------------------------------------------------------
// GraphSAGE (supervised) forward for MI455X / gfx1250.
// Heavy GEMMs run through v_wmma_f32_16x16x32_f16 (wave32 WMMA).
// Problem is gather-bandwidth bound (~310 MB of random feature reads);
// GEMMs (~1.6 GFLOP) are fed from f16 fragments pre-swizzled to a
// WMMA per-lane layout (contiguous k = 16*half + s) so the inner loop is 4 uint4 loads + 1 WMMA.
// ---------------------------------------------------------------------------

#define FEAT 602
#define KPAD 608            // FEAT padded to multiple of 32
#define HID  128
#define NCLS 41
#define B    512
#define S1N  25
#define S2N  10
#define MAXDEG 128

typedef __attribute__((ext_vector_type(16))) _Float16 v16h;
typedef __attribute__((ext_vector_type(8)))  float    v8f;

union FragH { uint4 q[2]; v16h v; };

__device__ __forceinline__ uint32_t hmix(uint32_t x) {
    x ^= x >> 16; x *= 0x7feb352du;
    x ^= x >> 15; x *= 0x846ca68bu;
    x ^= x >> 16; return x;
}

__device__ __forceinline__ uint32_t pack2h(float a, float b) {
    union { _Float16 h[2]; uint32_t u; } p;
    p.h[0] = (_Float16)a; p.h[1] = (_Float16)b;
    return p.u;
}

// --------------------------- neighbor sampling -----------------------------
// out[t] = adj[nodes[t/perNode]][hash(seed^t) & 127]
__global__ void k_sample(const int* __restrict__ nodes, int perNode,
                         const int* __restrict__ adjm,
                         int* __restrict__ out, int count, uint32_t seed) {
    int tid = blockIdx.x * blockDim.x + threadIdx.x;
    if (tid >= count) return;
    int node = nodes[tid / perNode];
    uint32_t col = hmix(seed ^ (uint32_t)tid) & (MAXDEG - 1);
    out[tid] = adjm[(size_t)node * MAXDEG + col];
}

// ----------------------- gather (+mean) to f16, K padded -------------------
// row i of dst[.,KPAD] = f16( mean_{n<nsamp} features[idx[i*nsamp+n], :] )
__global__ void k_gather_mean(const float* __restrict__ feat,
                              const int* __restrict__ idx, int nsamp,
                              _Float16* __restrict__ dst) {
    __shared__ int sidx[32];
    int i = blockIdx.x;
    if (threadIdx.x < nsamp) sidx[threadIdx.x] = idx[i * nsamp + threadIdx.x];
    __syncthreads();
    float inv = 1.0f / (float)nsamp;
    for (int f = threadIdx.x; f < KPAD; f += blockDim.x) {
        float acc = 0.0f;
        if (f < FEAT) {
            for (int n = 0; n < nsamp; ++n) {
                if (n + 1 < nsamp)
                    __builtin_prefetch(&feat[(size_t)sidx[n + 1] * FEAT + f], 0, 1);
                acc += feat[(size_t)sidx[n] * FEAT + f];
            }
        }
        dst[(size_t)i * KPAD + f] = (_Float16)(acc * inv);
    }
}

// ----------------- pack weights [K,128] f32 -> WMMA B fragments ------------
// Fragment layout (16x16x32 f16 B, 32x16): lane = 16*khalf + n,
// dword v (0..7) holds K = 32*t + 16*khalf + 2v, 2v+1 for column n of tile j.
// Stored as dst[((j*ksteps + t)*32 + lane)*8 + v] -> 8 contiguous dwords/lane.
__global__ void k_pack_w(const float* __restrict__ W, int Korig, int ksteps,
                         uint32_t* __restrict__ dst) {
    int tid = blockIdx.x * blockDim.x + threadIdx.x;
    int total = ksteps * 8 * 32 * 8;
    if (tid >= total) return;
    int v    = tid & 7;
    int lane = (tid >> 3) & 31;
    int q    = tid >> 8;          // j*ksteps + t
    int t    = q % ksteps;
    int j    = q / ksteps;
    int k0   = t * 32 + 16 * (lane >> 4) + 2 * v;
    int col  = j * 16 + (lane & 15);
    float a = (k0     < Korig) ? W[(size_t)k0 * HID + col]       : 0.0f;
    float b = (k0 + 1 < Korig) ? W[(size_t)(k0 + 1) * HID + col] : 0.0f;
    dst[tid] = pack2h(a, b);
}

// ------------------------------ WMMA GEMM ----------------------------------
// C[M,ldc] f32 (+= written at colOff) = A[M,lda] f16 row-major  x  Bpacked.
// One wave computes one 16x16 tile; 4 waves per block; no divergence around
// the WMMA (EXEC all-ones).  A fragment: per lane, two contiguous uint4 loads
// (f16 16x32 A layout: K = kbase + 16*half + [0..15]).
__global__ void k_gemm_wmma(const _Float16* __restrict__ A, int lda,
                            const uint32_t* __restrict__ Bp, int ksteps,
                            float* __restrict__ C, int ldc, int colOff,
                            int tilesN, int totalTiles, int doRelu) {
    int wave = threadIdx.x >> 5;
    int lane = threadIdx.x & 31;
    int tile = blockIdx.x * (blockDim.x >> 5) + wave;
    if (tile >= totalTiles) return;             // wave-uniform exit
    int tm = tile / tilesN;
    int tn = tile % tilesN;
    int half = lane >> 4;
    int mrow = tm * 16 + (lane & 15);
    const _Float16* arow = A + (size_t)mrow * lda + half * 16;
    const uint32_t* bptr = Bp + ((size_t)(tn * ksteps) * 32 + lane) * 8;

    v8f acc = {};
    for (int t = 0; t < ksteps; ++t) {
        FragH a, b;
        a.q[0] = *(const uint4*)(arow);
        a.q[1] = *(const uint4*)(arow + 8);
        b.q[0] = *(const uint4*)(bptr);
        b.q[1] = *(const uint4*)(bptr + 4);
        acc = __builtin_amdgcn_wmma_f32_16x16x32_f16(
            /*neg_a=*/false, a.v, /*neg_b=*/false, b.v,
            /*c_mod=*/(short)0, acc, /*reuse_a=*/false, /*reuse_b=*/false);
        arow += 32;
        bptr += 32 * 8;
    }
    int n = colOff + tn * 16 + (lane & 15);
    int m0 = tm * 16 + 8 * half;
    #pragma unroll
    for (int r = 0; r < 8; ++r) {
        float v = acc[r];
        if (doRelu) v = fmaxf(v, 0.0f);
        C[(size_t)(m0 + r) * ldc + n] = v;
    }
}

// ------------------- inter-layer mean / cast to f16 ------------------------
__global__ void k_mean_h1(const float* __restrict__ h1o, _Float16* __restrict__ dst) {
    int i = blockIdx.x, c = threadIdx.x;   // 512 x 256
    float acc = 0.0f;
    for (int n = 0; n < S2N; ++n)
        acc += h1o[(size_t)(i * S2N + n) * 256 + c];
    dst[(size_t)i * 256 + c] = (_Float16)(acc * (1.0f / S2N));
}

__global__ void k_cast_h(const float* __restrict__ src, _Float16* __restrict__ dst, int n) {
    int tid = blockIdx.x * blockDim.x + threadIdx.x;
    if (tid < n) dst[tid] = (_Float16)src[tid];
}

// ----------------- head: l2-normalize, dense(41), softmax ------------------
__global__ void k_head(const float* __restrict__ outb,
                       const float* __restrict__ Wp,
                       const float* __restrict__ bp,
                       float* __restrict__ pred) {
    __shared__ float row[256];
    __shared__ float red[256];
    __shared__ float logits[NCLS];
    int i = blockIdx.x, t = threadIdx.x;
    float v = outb[(size_t)i * 256 + t];
    row[t] = v;
    red[t] = v * v;
    __syncthreads();
    for (int s = 128; s > 0; s >>= 1) {
        if (t < s) red[t] += red[t + s];
        __syncthreads();
    }
    float scale = rsqrtf(fmaxf(red[0], 1e-12f));
    if (t < NCLS) {
        float acc = bp[t];
        for (int k = 0; k < 256; ++k)
            acc += row[k] * scale * Wp[(size_t)k * NCLS + t];
        logits[t] = acc;
    }
    __syncthreads();
    if (t == 0) {
        float mx = logits[0];
        for (int c = 1; c < NCLS; ++c) mx = fmaxf(mx, logits[c]);
        float s = 0.0f;
        for (int c = 0; c < NCLS; ++c) { float e = __expf(logits[c] - mx); logits[c] = e; s += e; }
        float inv = 1.0f / s;
        for (int c = 0; c < NCLS; ++c) pred[(size_t)i * NCLS + c] = logits[c] * inv;
    }
}

// ---------------------------------------------------------------------------
extern "C" void kernel_launch(void* const* d_in, const int* in_sizes, int n_in,
                              void* d_out, int out_size, void* d_ws, size_t ws_size,
                              hipStream_t stream) {
    (void)in_sizes; (void)n_in; (void)out_size; (void)ws_size;
    const float* features = (const float*)d_in[0];
    // d_in[1] = labels (unused in forward)
    const float* Ws0 = (const float*)d_in[2];
    const float* Wn0 = (const float*)d_in[3];
    const float* Ws1 = (const float*)d_in[4];
    const float* Wn1 = (const float*)d_in[5];
    const float* Wp  = (const float*)d_in[6];
    const float* bp  = (const float*)d_in[7];
    const int*   adjm  = (const int*)d_in[8];
    const int*   batch = (const int*)d_in[9];
    float* pred = (float*)d_out;

    // bump-allocate scratch out of d_ws (256B aligned)
    char* ws = (char*)d_ws;
    size_t off = 0;
    auto alloc = [&](size_t bytes) -> void* {
        off = (off + 255) & ~(size_t)255;
        void* p = ws + off;
        off += bytes;
        return p;
    };
    int*      s1    = (int*)     alloc((size_t)B * S2N * 4);                // 5120
    int*      s2    = (int*)     alloc((size_t)B * S2N * S1N * 4);          // 128000
    _Float16* A1    = (_Float16*)alloc((size_t)B * S2N * KPAD * 2);         // feat[s1]
    _Float16* A2    = (_Float16*)alloc((size_t)B * S2N * KPAD * 2);         // mean feat[s2]
    _Float16* A3    = (_Float16*)alloc((size_t)B * KPAD * 2);               // feat[s0]
    _Float16* A4    = (_Float16*)alloc((size_t)B * KPAD * 2);               // mean feat[s1]
    uint32_t* Ws0p  = (uint32_t*)alloc((size_t)19 * 8 * 32 * 8 * 4);
    uint32_t* Wn0p  = (uint32_t*)alloc((size_t)19 * 8 * 32 * 8 * 4);
    uint32_t* Ws1p  = (uint32_t*)alloc((size_t)8  * 8 * 32 * 8 * 4);
    uint32_t* Wn1p  = (uint32_t*)alloc((size_t)8  * 8 * 32 * 8 * 4);
    float*    h1o   = (float*)   alloc((size_t)B * S2N * 256 * 4);          // [5120,256]
    float*    h0o   = (float*)   alloc((size_t)B * 256 * 4);                // [512,256]
    _Float16* A5    = (_Float16*)alloc((size_t)B * 256 * 2);                // f16(h0o)
    _Float16* A6    = (_Float16*)alloc((size_t)B * 256 * 2);                // f16(mean h1o)
    float*    outb  = (float*)   alloc((size_t)B * 256 * 4);                // layer-1 out

    // 1) sampling (stand-in hash PRNG; deterministic per call)
    k_sample<<<20, 256, 0, stream>>>(batch, S2N, adjm, s1, B * S2N, 0x9E3779B9u);
    k_sample<<<500, 256, 0, stream>>>(s1, S1N, adjm, s2, B * S2N * S1N, 0x85EBCA6Bu);

    // 2) weight packing into WMMA fragment order
    k_pack_w<<<152, 256, 0, stream>>>(Ws0, FEAT, 19, Ws0p);
    k_pack_w<<<152, 256, 0, stream>>>(Wn0, FEAT, 19, Wn0p);
    k_pack_w<<<64,  256, 0, stream>>>(Ws1, 256, 8, Ws1p);
    k_pack_w<<<64,  256, 0, stream>>>(Wn1, 256, 8, Wn1p);

    // 3) gathers / means -> f16 activation matrices (bandwidth-dominant step)
    k_gather_mean<<<B * S2N, 256, 0, stream>>>(features, s1, 1,   A1);
    k_gather_mean<<<B * S2N, 256, 0, stream>>>(features, s2, S1N, A2);
    k_gather_mean<<<B,       256, 0, stream>>>(features, batch, 1, A3);
    k_gather_mean<<<B,       256, 0, stream>>>(features, s1, S2N, A4);

    // 4) layer 0 (relu, concat via colOff): h1o = [A1@Ws0 | A2@Wn0], h0o = [A3@Ws0 | A4@Wn0]
    k_gemm_wmma<<<640, 128, 0, stream>>>(A1, KPAD, Ws0p, 19, h1o, 256, 0,   8, 2560, 1);
    k_gemm_wmma<<<640, 128, 0, stream>>>(A2, KPAD, Wn0p, 19, h1o, 256, 128, 8, 2560, 1);
    k_gemm_wmma<<<64,  128, 0, stream>>>(A3, KPAD, Ws0p, 19, h0o, 256, 0,   8, 256, 1);
    k_gemm_wmma<<<64,  128, 0, stream>>>(A4, KPAD, Wn0p, 19, h0o, 256, 128, 8, 256, 1);

    // 5) inter-layer mean / cast
    k_mean_h1<<<B, 256, 0, stream>>>(h1o, A6);
    k_cast_h<<<(B * 256 + 255) / 256, 256, 0, stream>>>(h0o, A5, B * 256);

    // 6) layer 1 (identity act): outb = [A5@Ws1 | A6@Wn1]
    k_gemm_wmma<<<64, 128, 0, stream>>>(A5, 256, Ws1p, 8, outb, 256, 0,   8, 256, 0);
    k_gemm_wmma<<<64, 128, 0, stream>>>(A6, 256, Wn1p, 8, outb, 256, 128, 8, 256, 0);

    // 7) l2-normalize + node_pred dense + softmax
    k_head<<<B, 256, 0, stream>>>(outb, Wp, bp, pred);
}
